// CosineAttention_65377992179822
// MI455X (gfx1250) — matrix operand order repaired and checked by reading, not compiled, not run on hardware
//
#include <hip/hip_runtime.h>
#include <hip/hip_bf16.h>
#include <math.h>
#include <stdint.h>

typedef __attribute__((ext_vector_type(16))) _Float16 v16h;
typedef __attribute__((ext_vector_type(8)))  _Float16 h8;
typedef __attribute__((ext_vector_type(8)))  float    v8f;

#define B_    8
#define N_    4200
#define NF_   512
#define H_    8
#define D_    64
#define W_    35
#define WN_   120           // windows per batch
#define M_    33600         // B*N
#define PROJ_ 256
#define VD_   128
#define LDK   34            // padded LDS K-stride (halves) for 32-wide GEMM tiles
#define ALD   88            // padded LDS row stride (halves) in attention (176B, 16B aligned)

#if __has_builtin(__builtin_amdgcn_tensor_load_to_lds)
#define HAVE_TDM 1
#pragma message("CDNA5 probe: tensor_load_to_lds builtin AVAILABLE -> TDM path compiled")
#else
#pragma message("CDNA5 probe: tensor_load_to_lds builtin NOT available -> fallback loads")
#endif

// ---------------------------------------------------------------------------
// WMMA fragment loaders (per CDNA5 ISA 16-bit A/B VGPR layouts, wave32)
// ---------------------------------------------------------------------------
__device__ inline v16h load_a_frag(const _Float16* base, int stride, int row0,
                                   int k0, int lane) {
    int lm = lane & 15, half = lane >> 4;
    const _Float16* rp = base + (size_t)(row0 + lm) * stride;
    v16h a;
#pragma unroll
    for (int p = 0; p < 8; ++p) {
        int kb = k0 + ((p < 4) ? (2 * p + 8 * half) : (16 + 2 * (p - 4) + 8 * half));
        a[2 * p]     = rp[kb];
        a[2 * p + 1] = rp[kb + 1];
    }
    return a;
}

// B stored "transposed": Bs[n][k] (= weight[n][k]); fragment covers n-tile base nrow0
__device__ inline v16h load_b_frag(const _Float16* base, int stride, int nrow0,
                                   int k0, int lane) {
    int lm = lane & 15, half = lane >> 4;
    const _Float16* rp = base + (size_t)(nrow0 + lm) * stride;
    v16h b;
#pragma unroll
    for (int r = 0; r < 8; ++r) {
        int kb = k0 + ((r < 4) ? (2 * r + 8 * half) : (16 + 2 * (r - 4) + 8 * half));
        b[2 * r]     = rp[kb];
        b[2 * r + 1] = rp[kb + 1];
    }
    return b;
}

// ---------------------------------------------------------------------------
// Tensor Data Mover: 2D 35x64 f16 tile, global -> LDS, padded to ALD-half rows
// ---------------------------------------------------------------------------
#ifdef HAVE_TDM
typedef unsigned int u32x4 __attribute__((ext_vector_type(4)));
typedef int          i32x4 __attribute__((ext_vector_type(4)));
typedef int          i32x8 __attribute__((ext_vector_type(8)));

__device__ inline void tdm_load_tile_35x64_f16(const _Float16* gsrc, _Float16* lds_dst)
{
    uint64_t ga = (uint64_t)(uintptr_t)gsrc;
    u32x4 g0;
    g0[0] = 1u;                                     // count=1 (user descriptor)
    g0[1] = (uint32_t)(uintptr_t)lds_dst;           // lds_addr (low 32b of flat LDS addr)
    g0[2] = (uint32_t)ga;                           // global_addr[31:0]
    g0[3] = ((uint32_t)(ga >> 32) & 0x01FFFFFFu)    // global_addr[56:32]
          | (2u << 30);                             // type = 2 ("image")
    i32x8 g1;
    g1[0] = (int)((1u << 16)                        // data_size = 2 bytes
                | (1u << 20)                        // pad_enable
                | (4u << 22)                        // pad_interval: 32 DWORDs (one 64-half row)
                | (11u << 25));                     // pad_amount: 12 DWORDs -> 88-half stride
    g1[1] = (int)(64u << 16);                       // tensor_dim0[15:0] = 64
    g1[2] = (int)(35u << 16);                       // tensor_dim0[31:16]=0 | tensor_dim1[15:0]=35
    g1[3] = (int)(64u << 16);                       // tensor_dim1[31:16]=0 | tile_dim0 = 64
    g1[4] = (int)(35u);                             // tile_dim1 = 35 | tile_dim2 = 0
    g1[5] = (int)(64u);                             // tensor_dim0_stride[31:0] = 64
    g1[6] = 0;                                      // stride hi | tensor_dim1_stride lo
    g1[7] = 0;
    i32x4 gz = {};
#if __clang_major__ >= 23
    i32x8 gz8 = {};
    __builtin_amdgcn_tensor_load_to_lds(g0, g1, gz, gz, gz8, 0);
#else
    __builtin_amdgcn_tensor_load_to_lds(g0, g1, gz, gz, 0);
#endif
}
#endif

// ---------------------------------------------------------------------------
// Generic GEMM:  C[M,N] = A[M,K] * Wt[N,K]^T   (fp32 A/W, f16 WMMA, fp32 or f16 out)
// Block tile 64(M) x 128(N), 8 waves, each wave 16x64 (4 accumulators), BK=32.
// All six b128 staging loads issued before conversion/LDS stores so they
// pipeline through the memory system (partial loadcnt waits).
// mode 0: plain fp32 row-major C.   mode 1: f16 QKV scatter into q/k/v [B,w,H,W,D].
// Requires M%64==0, N%128==0, K%32==0 (true for every call here).
// ---------------------------------------------------------------------------
__global__ __launch_bounds__(256) void gemm_wmma_kernel(
    const float* __restrict__ A, const float* __restrict__ Bw,
    float* __restrict__ C, _Float16* __restrict__ Q, _Float16* __restrict__ Kp,
    _Float16* __restrict__ V, int M, int N, int K, int mode)
{
    __shared__ __attribute__((aligned(16))) _Float16 As[64][LDK];
    __shared__ __attribute__((aligned(16))) _Float16 Bs[128][LDK];

    const int tid  = threadIdx.x;
    const int lane = tid & 31;
    const int wave = tid >> 5;
    const int lm   = lane & 15;
    const int half = lane >> 4;
    const int wm   = wave & 3;      // 0..3 : 16-row strip
    const int wn   = wave >> 2;     // 0..1 : 64-col strip
    const int bm   = blockIdx.x * 64;
    const int bn   = blockIdx.y * 128;

    // Per-thread staging coordinates (fixed across K-steps)
    int ra[2], ca[2], rb[4], cb[4];
#pragma unroll
    for (int i = 0; i < 2; ++i) {
        int idx = tid + i * 256;                     // 0..511  (A: 64x32 = 512 float4)
        ra[i] = idx >> 3; ca[i] = (idx & 7) * 4;
    }
#pragma unroll
    for (int i = 0; i < 4; ++i) {
        int idx = tid + i * 256;                     // 0..1023 (B: 128x32 = 1024 float4)
        rb[i] = idx >> 3; cb[i] = (idx & 7) * 4;
    }

    v8f acc[4] = {};

    for (int k0 = 0; k0 < K; k0 += 32) {
        // --- issue ALL tile loads first (6 x global_load_b128 in flight)
        float4 fa[2], fb[4];
#pragma unroll
        for (int i = 0; i < 2; ++i)
            fa[i] = *reinterpret_cast<const float4*>(
                &A[(size_t)(bm + ra[i]) * K + k0 + ca[i]]);
#pragma unroll
        for (int i = 0; i < 4; ++i)
            fb[i] = *reinterpret_cast<const float4*>(
                &Bw[(size_t)(bn + rb[i]) * K + k0 + cb[i]]);
        if (k0 + 32 < K) {                           // pull next tiles toward WGP
            __builtin_prefetch(&A [(size_t)(bm + (tid & 63))  * K + k0 + 32], 0, 1);
            __builtin_prefetch(&Bw[(size_t)(bn + (tid & 127)) * K + k0 + 32], 0, 1);
        }
        // --- convert + LDS store (compiler retires loads with partial waits)
#pragma unroll
        for (int i = 0; i < 2; ++i) {
            As[ra[i]][ca[i] + 0] = (_Float16)fa[i].x;
            As[ra[i]][ca[i] + 1] = (_Float16)fa[i].y;
            As[ra[i]][ca[i] + 2] = (_Float16)fa[i].z;
            As[ra[i]][ca[i] + 3] = (_Float16)fa[i].w;
        }
#pragma unroll
        for (int i = 0; i < 4; ++i) {
            Bs[rb[i]][cb[i] + 0] = (_Float16)fb[i].x;
            Bs[rb[i]][cb[i] + 1] = (_Float16)fb[i].y;
            Bs[rb[i]][cb[i] + 2] = (_Float16)fb[i].z;
            Bs[rb[i]][cb[i] + 3] = (_Float16)fb[i].w;
        }
        __syncthreads();

        v16h a = load_a_frag(&As[0][0], LDK, wm * 16, 0, lane);
#pragma unroll
        for (int c = 0; c < 4; ++c) {
            v16h bb = load_b_frag(&Bs[0][0], LDK, wn * 64 + c * 16, 0, lane);
            acc[c] = __builtin_amdgcn_wmma_f32_16x16x32_f16(false, a, false, bb,
                                                            (short)0, acc[c], false, false);
        }
        __syncthreads();
    }

    if (mode == 0) {
#pragma unroll
        for (int v = 0; v < 8; ++v) {
            int row = bm + wm * 16 + v + 8 * half;
#pragma unroll
            for (int c = 0; c < 4; ++c)
                C[(size_t)row * N + bn + wn * 64 + c * 16 + lm] = acc[c][v];
        }
    } else {
        // col -> (h, d, s): col = h*192 + d*3 + s ; row -> (b*WN+w, n)
#pragma unroll
        for (int v = 0; v < 8; ++v) {
            int row  = bm + wm * 16 + v + 8 * half;
            int bw   = row / W_;
            int nwin = row % W_;
#pragma unroll
            for (int c = 0; c < 4; ++c) {
                int col = bn + wn * 64 + c * 16 + lm;
                float val = acc[c][v];
                int h = col / 192;
                int r = col % 192;
                int d = r / 3;
                int s = r % 3;
                size_t dst = (((size_t)bw * H_ + h) * W_ + nwin) * D_ + d;
                _Float16* p = (s == 0) ? Q : (s == 1) ? Kp : V;
                p[dst] = (_Float16)val;
            }
        }
    }
}

// ---------------------------------------------------------------------------
// L2-normalize q (scaled by temperature) and k, per (b,w,h,n) row of 64 (f16).
// ---------------------------------------------------------------------------
__global__ void normalize_qk_kernel(_Float16* __restrict__ q, _Float16* __restrict__ k,
                                    const float* __restrict__ temp, int rows)
{
    int r = blockIdx.x * blockDim.x + threadIdx.x;
    if (r >= rows) return;
    float t = *temp;
    _Float16* qp = q + (size_t)r * D_;
    _Float16* kp = k + (size_t)r * D_;
    float sq = 0.f, sk = 0.f;
#pragma unroll 4
    for (int d = 0; d < D_; ++d) {
        float qv = (float)qp[d], kv = (float)kp[d];
        sq += qv * qv; sk += kv * kv;
    }
    float iq = t / fmaxf(sqrtf(sq), 1e-12f);
    float ik = 1.0f / fmaxf(sqrtf(sk), 1e-12f);
#pragma unroll 4
    for (int d = 0; d < D_; ++d) {
        qp[d] = (_Float16)((float)qp[d] * iq);
        kp[d] = (_Float16)((float)kp[d] * ik);
    }
}

// ---------------------------------------------------------------------------
// Windowed attention, one block per (b, w, h).  35 padded to 48 rows / 64 K.
// Q/K tiles staged via TDM (tensor_load_to_lds) when available; QK^T and P*V
// via f16 WMMA; softmax + pos_bias + mask in fp32 LDS.
// ---------------------------------------------------------------------------
__global__ __launch_bounds__(128) void attn_kernel(
    const _Float16* __restrict__ Q, const _Float16* __restrict__ Km,
    const _Float16* __restrict__ V, const float* __restrict__ pos_bias,
    const unsigned char* __restrict__ mask, float* __restrict__ out)
{
    __shared__ __attribute__((aligned(16))) _Float16 Qs[48][ALD];
    __shared__ __attribute__((aligned(16))) _Float16 Ks[48][ALD];
    __shared__ __attribute__((aligned(16))) _Float16 Vt[64][ALD];   // Vt[d][j]
    __shared__ __attribute__((aligned(16))) _Float16 Ps[48][ALD];
    __shared__ float    Ss[48][49];
    __shared__ unsigned char mrow[48];

    const int tid  = threadIdx.x;
    const int lane = tid & 31;
    const int wave = tid >> 5;
    const int lm   = lane & 15;
    const int half = lane >> 4;

    const int bwh = blockIdx.x;
    const int h   = bwh & 7;
    const int bw  = bwh >> 3;            // b*WN + w
    const int b   = bw / WN_;
    const int wi  = bw % WN_;
    const size_t base = (size_t)bwh * W_ * D_;

#ifdef HAVE_TDM
    if (wave == 0) tdm_load_tile_35x64_f16(Q  + base, &Qs[0][0]);
    if (wave == 1) tdm_load_tile_35x64_f16(Km + base, &Ks[0][0]);
#else
    for (int idx = tid; idx < W_ * 8; idx += 128) {   // 8-half vector copies
        int r = idx >> 3, c8 = (idx & 7) * 8;
        *(h8*)&Qs[r][c8] = *(const h8*)&Q [base + r * D_ + c8];
        *(h8*)&Ks[r][c8] = *(const h8*)&Km[base + r * D_ + c8];
    }
#endif
    // zero the padded tail rows 35..47 (full padded width)
    for (int idx = tid; idx < 13 * ALD; idx += 128) {
        int r = 35 + idx / ALD, c = idx % ALD;
        Qs[r][c] = (_Float16)0.f;
        Ks[r][c] = (_Float16)0.f;
    }
    // V transposed into Vt[d][j], zero-padded j in [35,64)
    for (int idx = tid; idx < 64 * 8; idx += 128) {
        int r = idx >> 3, c8 = (idx & 7) * 8;         // r = j, c8 = d chunk
        h8 v8 = {};
        if (r < W_) v8 = *(const h8*)&V[base + r * D_ + c8];
#pragma unroll
        for (int e = 0; e < 8; ++e) Vt[c8 + e][r] = v8[e];
    }
    if (tid < 48)
        mrow[tid] = (tid < W_) ? mask[(size_t)b * N_ + wi * W_ + tid] : 1;
#ifdef HAVE_TDM
#if __has_builtin(__builtin_amdgcn_s_wait_tensorcnt)
    if (wave < 2) __builtin_amdgcn_s_wait_tensorcnt(0);
#endif
#endif
    __syncthreads();

    // ---- S = Q K^T : 9 tiles of 16x16, K=64 in two 32-steps
    for (int t = wave; t < 9; t += 4) {
        int ti = t / 3, tj = t % 3;
        v8f acc = {};
#pragma unroll
        for (int k0 = 0; k0 < 64; k0 += 32) {
            v16h a  = load_a_frag(&Qs[0][0], ALD, ti * 16, k0, lane);
            v16h bb = load_b_frag(&Ks[0][0], ALD, tj * 16, k0, lane);
            acc = __builtin_amdgcn_wmma_f32_16x16x32_f16(false, a, false, bb,
                                                         (short)0, acc, false, false);
        }
#pragma unroll
        for (int v = 0; v < 8; ++v)
            Ss[ti * 16 + v + 8 * half][tj * 16 + lm] = acc[v];
    }
    __syncthreads();

    // ---- softmax rows (bias + mask), write P (f16, zero-padded to 48x64)
    const float NEGBIG = -3.402823466e38f;
    if (tid < 48) {
        int r = tid;
        if (r < W_) {
            float vals[W_];
            float mx = NEGBIG;
            bool mi = mrow[r] != 0;
#pragma unroll 1
            for (int j = 0; j < W_; ++j) {
                float s = (mi || mrow[j]) ? NEGBIG
                        : Ss[r][j] + pos_bias[(size_t)h * W_ * W_ + r * W_ + j];
                vals[j] = s;
                mx = fmaxf(mx, s);
            }
            float sum = 0.f;
#pragma unroll 1
            for (int j = 0; j < W_; ++j) { float e = __expf(vals[j] - mx); vals[j] = e; sum += e; }
            float inv = 1.0f / sum;
#pragma unroll 1
            for (int j = 0; j < 64; ++j)
                Ps[r][j] = (_Float16)((j < W_) ? vals[j] * inv : 0.f);
        } else {
#pragma unroll 1
            for (int j = 0; j < 64; ++j) Ps[r][j] = (_Float16)0.f;
        }
    }
    __syncthreads();

    // ---- O = P V : 3 (i) x 4 (d) tiles, K=64 in two 32-steps
    for (int t = wave; t < 12; t += 4) {
        int ti = t >> 2, td = t & 3;
        v8f acc = {};
#pragma unroll
        for (int k0 = 0; k0 < 64; k0 += 32) {
            v16h a  = load_a_frag(&Ps[0][0], ALD, ti * 16, k0, lane);
            v16h bb = load_b_frag(&Vt[0][0], ALD, td * 16, k0, lane);
            acc = __builtin_amdgcn_wmma_f32_16x16x32_f16(false, a, false, bb,
                                                         (short)0, acc, false, false);
        }
#pragma unroll
        for (int v = 0; v < 8; ++v) {
            int i = ti * 16 + v + 8 * half;
            if (i < W_)
                out[((size_t)bw * W_ + i) * NF_ + h * D_ + td * 16 + lm] = acc[v];
        }
    }
}

// ---------------------------------------------------------------------------
// Window sums of `out` (masked) + inverse lengths, one block per (b,w).
// ---------------------------------------------------------------------------
__global__ __launch_bounds__(256) void reduce_kernel(
    const float* __restrict__ out, const unsigned char* __restrict__ mask,
    float* __restrict__ ssum, float* __restrict__ invlen)
{
    int bw = blockIdx.x;
    int b = bw / WN_, w = bw % WN_;
    if (threadIdx.x == 0) {
        int c = 0;
        for (int n = 0; n < W_; ++n)
            c += mask[(size_t)b * N_ + w * W_ + n] ? 0 : 1;
        invlen[bw] = 1.0f / ((c == 0) ? 1.0f : (float)c);
    }
    for (int c = threadIdx.x; c < NF_; c += 256) {
        float s = 0.f;
        for (int n = 0; n < W_; ++n)
            if (!mask[(size_t)b * N_ + w * W_ + n])
                s += out[((size_t)bw * W_ + n) * NF_ + c];
        ssum[(size_t)bw * NF_ + c] = s;
    }
}

// ---------------------------------------------------------------------------
// we = relu((ssum @ li_w^T) / len); weq = we @ q_w^T.  One block per (b,w).
// ---------------------------------------------------------------------------
__global__ __launch_bounds__(256) void weq_kernel(
    const float* __restrict__ ssum, const float* __restrict__ invlen,
    const float* __restrict__ li_w, const float* __restrict__ q_w,
    float* __restrict__ weq)
{
    int bw = blockIdx.x;
    __shared__ float srow[NF_];
    __shared__ float we[PROJ_];
    for (int c = threadIdx.x; c < NF_; c += 256)
        srow[c] = ssum[(size_t)bw * NF_ + c];
    __syncthreads();
    int c = threadIdx.x;                 // 256 == PROJ_
    float s = 0.f;
    for (int k = 0; k < NF_; ++k) s += srow[k] * li_w[(size_t)c * NF_ + k];
    we[c] = fmaxf(s * invlen[bw], 0.0f);
    __syncthreads();
    float s2 = 0.f;
    for (int k = 0; k < PROJ_; ++k) s2 += we[k] * q_w[(size_t)c * PROJ_ + k];
    weq[(size_t)bw * PROJ_ + c] = s2;
}

// ---------------------------------------------------------------------------
// kk += sinusoidal positional embedding (per in-window position)
// ---------------------------------------------------------------------------
__global__ void sinu_kernel(float* __restrict__ kk, const float* __restrict__ pos_scale)
{
    size_t idx = (size_t)blockIdx.x * 256 + threadIdx.x;   // M_ * 256 elems
    int m = (int)(idx >> 8);
    int c = (int)(idx & 255);
    int n = m % W_;
    int i = (c < 128) ? c : c - 128;
    float invf = __expf(-((float)(2 * i) / 256.0f) * 9.210340371976184f); // 10000^(-2i/256)
    float ang  = (float)n * invf;
    float v    = (c < 128) ? __sinf(ang) : __cosf(ang);
    kk[idx] += v * pos_scale[0];
}

// ---------------------------------------------------------------------------
// Second-stage attention: d2 = weq . kk / 16, softmax, o2 = a2 @ vv.
// ---------------------------------------------------------------------------
__global__ __launch_bounds__(128) void attn2_kernel(
    const float* __restrict__ weq, const float* __restrict__ kk,
    const float* __restrict__ vv, const unsigned char* __restrict__ mask,
    float* __restrict__ o2)
{
    int bw = blockIdx.x;
    int b = bw / WN_, w = bw % WN_;
    __shared__ float wq[PROJ_];
    __shared__ float prob[W_];
    for (int c = threadIdx.x; c < PROJ_; c += 128)
        wq[c] = weq[(size_t)bw * PROJ_ + c];
    __syncthreads();
    int j = threadIdx.x;
    if (j < W_) {
        const float* kr = kk + ((size_t)bw * W_ + j) * PROJ_;
        float s = 0.f;
        for (int k = 0; k < PROJ_; ++k) s += wq[k] * kr[k];
        s *= 0.0625f;                                   // 256^-0.5
        if (mask[(size_t)b * N_ + w * W_ + j]) s = -3.402823466e38f;
        prob[j] = s;
    }
    __syncthreads();
    if (threadIdx.x == 0) {
        float mx = -3.402823466e38f;
        for (int t = 0; t < W_; ++t) mx = fmaxf(mx, prob[t]);
        float sm = 0.f;
        for (int t = 0; t < W_; ++t) { float e = __expf(prob[t] - mx); prob[t] = e; sm += e; }
        float inv = 1.0f / sm;
        for (int t = 0; t < W_; ++t) prob[t] *= inv;
    }
    __syncthreads();
    int d = threadIdx.x;
    if (d < VD_) {
        float s = 0.f;
        for (int jj = 0; jj < W_; ++jj)
            s += prob[jj] * vv[((size_t)bw * W_ + jj) * VD_ + d];
        o2[(size_t)bw * VD_ + d] = s;
    }
}

// ---------------------------------------------------------------------------
// feat = concat(out[.,512], shifted o2 broadcast [.,128]) -> [M, 640]
// ---------------------------------------------------------------------------
__global__ void feat_kernel(const float* __restrict__ out, const float* __restrict__ o2,
                            const float* __restrict__ bos, float* __restrict__ feat)
{
    size_t idx = (size_t)blockIdx.x * 256 + threadIdx.x;
    if (idx >= (size_t)M_ * 640) return;
    int m = (int)(idx / 640);
    int c = (int)(idx % 640);
    float v;
    if (c < NF_) {
        v = out[(size_t)m * NF_ + c];
    } else {
        int d  = c - NF_;
        int bw = m / W_;
        int wi = bw % WN_;
        v = (wi == 0) ? bos[d] : o2[(size_t)(bw - 1) * VD_ + d];
    }
    feat[idx] = v;
}

// ---------------------------------------------------------------------------
extern "C" void kernel_launch(void* const* d_in, const int* in_sizes, int n_in,
                              void* d_out, int out_size, void* d_ws, size_t ws_size,
                              hipStream_t stream)
{
    (void)in_sizes; (void)n_in; (void)out_size; (void)ws_size;

    const float*         x     = (const float*)d_in[0];
    const unsigned char* mask  = (const unsigned char*)d_in[1];
    const float*         pbias = (const float*)d_in[2];
    const float*         temp  = (const float*)d_in[3];
    const float*         qkv_w = (const float*)d_in[4];
    const float*         out_w = (const float*)d_in[5];
    const float*         li_w  = (const float*)d_in[6];
    const float*         q_w   = (const float*)d_in[7];
    const float*         k_w   = (const float*)d_in[8];
    const float*         v_w   = (const float*)d_in[9];
    const float*         bos   = (const float*)d_in[10];
    const float*         pscal = (const float*)d_in[11];
    float*               y     = (float*)d_out;

    // Workspace layout (f16 q/k/v, then fp32 buffers)
    _Float16* qT   = (_Float16*)d_ws;                          // M*512 h
    _Float16* kT   = qT + (size_t)M_ * NF_;                    // M*512 h
    _Float16* vT   = kT + (size_t)M_ * NF_;                    // M*512 h
    float*    aout = (float*)(vT + (size_t)M_ * NF_);          // M*512 f
    float*    ssum = aout + (size_t)M_ * NF_;                  // 960*512
    float*    ivl  = ssum + (size_t)B_ * WN_ * NF_;            // 960
    float*    weq  = ivl  + (size_t)B_ * WN_;                  // 960*256
    float*    o2   = weq  + (size_t)B_ * WN_ * PROJ_;          // 960*128
    // aliases (lifetimes disjoint, stream-ordered):
    float*    kk   = (float*)qT;                               // M*256 f (q dead after attn)
    float*    vv   = kk + (size_t)M_ * PROJ_;                  // M*128 f
    float*    feat = (float*)qT;                               // M*640 f (after attn2)

    // 1) QKV projection -> scattered f16 q/k/v [B,w,H,W,D]
    gemm_wmma_kernel<<<dim3(M_ / 64, 1536 / 128), 256, 0, stream>>>(
        x, qkv_w, nullptr, qT, kT, vT, M_, 1536, NF_, 1);

    // 2) l2norm q (x temperature) and k
    normalize_qk_kernel<<<(M_ * H_) / 256, 256, 0, stream>>>(qT, kT, temp, M_ * H_);

    // 3) windowed cosine attention (TDM-staged Q/K tiles)
    attn_kernel<<<B_ * WN_ * H_, 128, 0, stream>>>(qT, kT, vT, pbias, mask, aout);

    // 4) window sums + lengths
    reduce_kernel<<<B_ * WN_, 256, 0, stream>>>(aout, mask, ssum, ivl);

    // 5) we -> we_q
    weq_kernel<<<B_ * WN_, 256, 0, stream>>>(ssum, ivl, li_w, q_w, weq);

    // 6) vv = out @ v_w^T ; 7) kk = out @ k_w^T   (write into dead q region)
    gemm_wmma_kernel<<<dim3(M_ / 64, VD_ / 128), 256, 0, stream>>>(
        aout, v_w, vv, nullptr, nullptr, nullptr, M_, VD_, NF_, 0);
    gemm_wmma_kernel<<<dim3(M_ / 64, PROJ_ / 128), 256, 0, stream>>>(
        aout, k_w, kk, nullptr, nullptr, nullptr, M_, PROJ_, NF_, 0);

    // 8) kk += sinusoidal embedding
    sinu_kernel<<<M_, 256, 0, stream>>>(kk, pscal);

    // 9) second-stage window attention -> o2
    attn2_kernel<<<B_ * WN_, 128, 0, stream>>>(weq, kk, vv, mask, o2);

    // 10) feat = [out, shift(o2)]  (reuses q/k/v span after attn2 completes)
    feat_kernel<<<(int)(((size_t)M_ * 640 + 255) / 256), 256, 0, stream>>>(
        aout, o2, bos, feat);

    // 11) y = feat @ out_w^T
    gemm_wmma_kernel<<<dim3(M_ / 64, NF_ / 128), 256, 0, stream>>>(
        feat, out_w, y, nullptr, nullptr, nullptr, M_, NF_, 640, 0);
}
